// HFOpenMoeAttention_64089501991566
// MI455X (gfx1250) — matrix operand order repaired and checked by reading, not compiled, not run on hardware
//
#include <hip/hip_runtime.h>

#define B_   2
#define S_   2048
#define HID_ 2048
#define NH_  16
#define NKV_ 4
#define HD_  128

typedef __bf16 bf16;
typedef __attribute__((ext_vector_type(16))) __bf16 v16bf;
typedef __attribute__((ext_vector_type(8)))  float  v8f;
typedef __attribute__((ext_vector_type(4)))  unsigned int u32x4;

// ---------------- bf16 helpers (round-to-nearest-even) ----------------
static __device__ __forceinline__ bf16 f2bf(float f) {
  unsigned u = __builtin_bit_cast(unsigned, f);
  unsigned r = u + 0x7FFFu + ((u >> 16) & 1u);
  unsigned short h = (unsigned short)(r >> 16);
  return __builtin_bit_cast(bf16, h);
}
static __device__ __forceinline__ float bf2f(bf16 b) {
  unsigned short h = __builtin_bit_cast(unsigned short, b);
  unsigned u = ((unsigned)h) << 16;
  return __builtin_bit_cast(float, u);
}

union FragU { v16bf v; u32x4 q[2]; };

// A-operand fragment (16x32 bf16), ISA 7.12.2 "16-bit A-Matrix 16x32":
// lanes 0-15: row M=lane, K {0..7, 16..23}; lanes 16-31: row M=lane-16, K {8..15, 24..31}.
static __device__ __forceinline__ v16bf load_frag_a(const bf16* base, int pitch,
                                                    int row0, int k0, int lane) {
  int row = row0 + (lane & 15);
  int kb  = k0 + ((lane >> 4) << 3);               // +0 or +8
  const bf16* p = base + (size_t)row * pitch + kb;
  FragU f;
  f.q[0] = *reinterpret_cast<const u32x4*>(p);      // K kb..kb+7
  f.q[1] = *reinterpret_cast<const u32x4*>(p + 16); // K kb+16..kb+23
  return f.v;
}

// B-operand fragment (32x16 bf16) loaded from B^T rows (row n, K contiguous):
// lanes 0-15 hold K {0..7, 16..23}, lanes 16-31 hold K {8..15, 24..31}.
static __device__ __forceinline__ v16bf load_frag_b(const bf16* base, int pitch,
                                                    int n0, int k0, int lane) {
  int n  = n0 + (lane & 15);
  int kb = k0 + ((lane >> 4) << 3);                // +0 or +8
  const bf16* p = base + (size_t)n * pitch + kb;
  FragU f;
  f.q[0] = *reinterpret_cast<const u32x4*>(p);     // K kb..kb+7
  f.q[1] = *reinterpret_cast<const u32x4*>(p + 16); // K kb+16..kb+23
  return f.v;
}

static __device__ __forceinline__ v8f wmma_bf(v16bf a, v16bf b, v8f c) {
  return __builtin_amdgcn_wmma_f32_16x16x32_bf16(false, a, false, b, (short)0, c,
                                                 false, false);
}

// ---------------- fp32 -> bf16 conversion ----------------
__global__ void cvt_f32_bf16_kernel(const float* __restrict__ src,
                                    bf16* __restrict__ dst, long n) {
  long i = (long)blockIdx.x * blockDim.x + threadIdx.x;
  long stride = (long)gridDim.x * blockDim.x;
  for (; i < n; i += stride) dst[i] = f2bf(src[i]);
}

// ---------------- C[M,N] = A[M,K] @ W[N,K]^T, bf16 WMMA, f32 accum ----------------
// Block = 256 threads = 8 waves; block tile 256(M) x 128(N); wave tile 64x64 (4x4).
// B tile (128 rows x 32 k) staged in LDS via global_load_async_to_lds_b128,
// double-buffered on ASYNCcnt; A streamed straight from global (L2-resident).
#define BPITCH 40   // bf16 elems per LDS B row: 80 bytes, 16B aligned, bank stride 20
__global__ __launch_bounds__(256) void gemm_bf16_kernel(
    const bf16* __restrict__ A, const bf16* __restrict__ W,
    float* __restrict__ Cf, bf16* __restrict__ Cb, int M, int N, int K) {
  __shared__ bf16 Bs[2][128][BPITCH];
  int lane = threadIdx.x & 31;
  int wave = threadIdx.x >> 5;
  int wm = wave >> 1;                       // 0..3 -> M
  int wn = wave & 1;                        // 0..1 -> N
  int row0 = blockIdx.y * 256 + wm * 64;
  int colB = blockIdx.x * 128;              // block N base (global W row)
  int colW = wn * 64;                       // wave N offset inside block

  // async-copy mapping: 256 threads x 2 b128 = 128 rows x 64B (32 bf16 of K)
  int crow   = threadIdx.x >> 1;            // 0..127
  int cchunk = (threadIdx.x & 1) * 2;       // b128 chunk pair {0,1} or {2,3}
  const bf16* wrow = W + (size_t)(colB + crow) * K + cchunk * 8;

  v8f acc[4][4];
#pragma unroll
  for (int i = 0; i < 4; ++i)
#pragma unroll
    for (int j = 0; j < 4; ++j) acc[i][j] = (v8f){0.f,0.f,0.f,0.f,0.f,0.f,0.f,0.f};

  int nsteps = K / 32;
  // issue stage 0 into buffer 0
  {
    unsigned d0 = (unsigned)(size_t)&Bs[0][crow][cchunk * 8];
    unsigned long long s0 = (unsigned long long)(size_t)wrow;
    asm volatile("global_load_async_to_lds_b128 %0, %1, off"
                 :: "v"(d0), "v"(s0) : "memory");
    asm volatile("global_load_async_to_lds_b128 %0, %1, off"
                 :: "v"(d0 + 16u), "v"(s0 + 16ull) : "memory");
  }

  for (int s = 0; s < nsteps; ++s) {
    int k0 = s << 5;
    if (s + 1 < nsteps) {                   // prefetch next B tile into other buffer
      unsigned d1 = (unsigned)(size_t)&Bs[(s + 1) & 1][crow][cchunk * 8];
      unsigned long long s1 = (unsigned long long)(size_t)(wrow + k0 + 32);
      asm volatile("global_load_async_to_lds_b128 %0, %1, off"
                   :: "v"(d1), "v"(s1) : "memory");
      asm volatile("global_load_async_to_lds_b128 %0, %1, off"
                   :: "v"(d1 + 16u), "v"(s1 + 16ull) : "memory");
      asm volatile("s_wait_asynccnt 0x2" ::: "memory");  // stage s done, s+1 in flight
    } else {
      asm volatile("s_wait_asynccnt 0x0" ::: "memory");
    }
    __syncthreads();                        // stage-s B tile visible to all waves

    if (k0 + 128 < K)                       // L2 prefetch hint for A
      __builtin_prefetch(A + (size_t)(row0 + (lane & 15)) * K + k0 + 128, 0, 1);

    v16bf a[4];
#pragma unroll
    for (int i = 0; i < 4; ++i) a[i] = load_frag_a(A, K, row0 + i * 16, k0, lane);
    const bf16* bs = &Bs[s & 1][0][0];
#pragma unroll
    for (int j = 0; j < 4; ++j) {
      v16bf b = load_frag_b(bs, BPITCH, colW + j * 16, 0, lane);
#pragma unroll
      for (int i = 0; i < 4; ++i) acc[i][j] = wmma_bf(a[i], b, acc[i][j]);
    }
    __syncthreads();                        // before buffer (s&1) is overwritten
  }

  // D layout: VGPR r, lanes 0-15 -> M=r, lanes 16-31 -> M=r+8; N = lane&15
  int g = lane >> 4, n = lane & 15;
  if (Cf) {
#pragma unroll
    for (int i = 0; i < 4; ++i)
#pragma unroll
      for (int j = 0; j < 4; ++j)
#pragma unroll
        for (int r = 0; r < 8; ++r) {
          int row = row0 + i * 16 + r + 8 * g;
          int col = colB + colW + j * 16 + n;
          Cf[(size_t)row * N + col] = acc[i][j][r];
        }
  } else {
#pragma unroll
    for (int i = 0; i < 4; ++i)
#pragma unroll
      for (int j = 0; j < 4; ++j)
#pragma unroll
        for (int r = 0; r < 8; ++r) {
          int row = row0 + i * 16 + r + 8 * g;
          int col = colB + colW + j * 16 + n;
          Cb[(size_t)row * N + col] = f2bf(acc[i][j][r]);
        }
  }
}

// ---------------- in-place RoPE on bf16 [B*S, heads*128] ----------------
__global__ void rope_kernel(bf16* __restrict__ x, const float* __restrict__ sinT,
                            const float* __restrict__ cosT, int heads, long total) {
  long i = (long)blockIdx.x * blockDim.x + threadIdx.x;
  if (i >= total) return;
  int d = (int)(i & 63);
  int h = (int)((i >> 6) % heads);
  long row = i / (64L * heads);
  int s = (int)(row & (S_ - 1));
  bf16* p = x + row * ((size_t)heads * HD_) + h * HD_;
  float x1 = bf2f(p[d]), x2 = bf2f(p[d + 64]);
  float c  = cosT[s * HD_ + d];   // cos[:,d] == cos[:,d+64] (tables are concatenated)
  float sn = sinT[s * HD_ + d];
  p[d]      = f2bf(x1 * c - x2 * sn);
  p[d + 64] = f2bf(x2 * c + x1 * sn);
}

// ---------------- V [B,S,NKV,HD] -> Vt [B,NKV,HD,S] ----------------
__global__ void vtrans_kernel(const bf16* __restrict__ V, bf16* __restrict__ Vt) {
  long i = (long)blockIdx.x * blockDim.x + threadIdx.x;
  long total = (long)B_ * S_ * NKV_ * HD_;
  if (i >= total) return;
  int d  = (int)(i & (HD_ - 1));
  int kv = (int)((i >> 7) & (NKV_ - 1));
  long bs = i >> 9;
  int s = (int)(bs & (S_ - 1));
  int b = (int)(bs >> 11);
  Vt[(((size_t)b * NKV_ + kv) * HD_ + d) * S_ + s] = V[i];
}

// ---------------- causal flash attention, one wave per 16-query tile ----------------
// Q [B,S,NH*HD] bf16 (roped), K [B,S,NKV*HD] bf16 (roped), Vt [B,NKV,HD,S] bf16.
// NOTE: reference applies NO 1/sqrt(d) scaling; mask col-0 fixup is a no-op on tril.
__global__ __launch_bounds__(256) void flash_attn_kernel(
    const bf16* __restrict__ Q, const bf16* __restrict__ K,
    const bf16* __restrict__ Vt, bf16* __restrict__ O) {
  __shared__ bf16 ldsP[8][16][40];  // per-wave 16x32 P tile, pitch 40 (16B-aligned rows)
  int lane = threadIdx.x & 31;
  int wave = threadIdx.x >> 5;
  int t  = blockIdx.x * 8 + wave;        // tile id in [0, B*NH*S/16)
  int b  = t >> 11;                      // / (NH * S/16) = /2048
  int h  = (t >> 7) & (NH_ - 1);
  int q0 = (t & 127) << 4;
  int kv = h >> 2;                       // GQA: n_rep = 4
  const bf16* Qb  = Q  + (size_t)b * S_ * (NH_ * HD_)  + h  * HD_;
  const bf16* Kb  = K  + (size_t)b * S_ * (NKV_ * HD_) + kv * HD_;
  const bf16* Vtb = Vt + ((size_t)(b * NKV_ + kv) * HD_) * S_;

  v16bf qf[4];
#pragma unroll
  for (int c = 0; c < 4; ++c) qf[c] = load_frag_a(Qb, NH_ * HD_, q0, c * 32, lane);

  v8f o[8];
#pragma unroll
  for (int dt = 0; dt < 8; ++dt) o[dt] = (v8f){0.f,0.f,0.f,0.f,0.f,0.f,0.f,0.f};
  float mrow[8], lrow[8];
#pragma unroll
  for (int r = 0; r < 8; ++r) { mrow[r] = -3.0e38f; lrow[r] = 0.f; }

  int g = lane >> 4, n = lane & 15;
  bf16 (*P)[40] = ldsP[wave];
  int nblk = (q0 + 16 + 31) >> 5;        // causal: only key blocks <= diag

  for (int kb = 0; kb < nblk; ++kb) {
    int kbase = kb << 5;
    v8f s0 = (v8f){0.f,0.f,0.f,0.f,0.f,0.f,0.f,0.f};
    v8f s1 = s0;
#pragma unroll
    for (int c = 0; c < 4; ++c) {        // S = Q @ K^T (k-dim = head dim, 4 chunks)
      v16bf kf0 = load_frag_b(Kb, NKV_ * HD_, kbase,      c * 32, lane);
      s0 = wmma_bf(qf[c], kf0, s0);
    }
#pragma unroll
    for (int c = 0; c < 4; ++c) {
      v16bf kf1 = load_frag_b(Kb, NKV_ * HD_, kbase + 16, c * 32, lane);
      s1 = wmma_bf(qf[c], kf1, s1);
    }
    // online softmax; row m = r + 8*g lives across the 16-lane half-wave
#pragma unroll
    for (int r = 0; r < 8; ++r) {
      int qrow = q0 + r + 8 * g;
      float x0 = (kbase + n      <= qrow) ? s0[r] : -3.0e38f;
      float x1 = (kbase + 16 + n <= qrow) ? s1[r] : -3.0e38f;
      float mx = fmaxf(x0, x1);
#pragma unroll
      for (int off = 1; off < 16; off <<= 1) mx = fmaxf(mx, __shfl_xor(mx, off, 32));
      float mnew = fmaxf(mrow[r], mx);
      float scl  = __expf(mrow[r] - mnew);
      float p0 = __expf(x0 - mnew);
      float p1 = __expf(x1 - mnew);
      float sum = p0 + p1;
#pragma unroll
      for (int off = 1; off < 16; off <<= 1) sum += __shfl_xor(sum, off, 32);
      lrow[r] = lrow[r] * scl + sum;
      mrow[r] = mnew;
#pragma unroll
      for (int dt = 0; dt < 8; ++dt) o[dt][r] *= scl;
      int m = r + 8 * g;
      P[m][n]      = f2bf(p0);           // re-layout D->A through LDS
      P[m][n + 16] = f2bf(p1);
    }
    asm volatile("s_wait_dscnt 0" ::: "memory");
    v16bf pa = load_frag_a(&P[0][0], 40, 0, 0, lane);
#pragma unroll
    for (int dt = 0; dt < 8; ++dt) {     // O += P @ V  (B operand from Vt rows)
      v16bf vf = load_frag_b(Vtb, S_, dt * 16, kbase, lane);
      o[dt] = wmma_bf(pa, vf, o[dt]);
    }
  }

  float inv[8];
#pragma unroll
  for (int r = 0; r < 8; ++r) inv[r] = 1.0f / lrow[r];
  bf16* Ob = O + (size_t)b * S_ * (NH_ * HD_) + h * HD_;
#pragma unroll
  for (int dt = 0; dt < 8; ++dt)
#pragma unroll
    for (int r = 0; r < 8; ++r) {
      int row = q0 + r + 8 * g;
      Ob[(size_t)row * (NH_ * HD_) + dt * 16 + n] = f2bf(o[dt][r] * inv[r]);
    }
}

// ---------------- host orchestration ----------------
extern "C" void kernel_launch(void* const* d_in, const int* in_sizes, int n_in,
                              void* d_out, int out_size, void* d_ws, size_t ws_size,
                              hipStream_t stream) {
  (void)in_sizes; (void)n_in; (void)out_size; (void)ws_size;
  const float* X    = (const float*)d_in[0];
  // d_in[1] (attention_mask) intentionally unused: tril causal + col0-fixup == causal
  const float* Wq   = (const float*)d_in[2];
  const float* Wk   = (const float*)d_in[3];
  const float* Wv   = (const float*)d_in[4];
  const float* Wo   = (const float*)d_in[5];
  const float* sinT = (const float*)d_in[6];
  const float* cosT = (const float*)d_in[7];

  const size_t M = (size_t)B_ * S_;           // 4096
  bf16* ws  = (bf16*)d_ws;
  bf16* Xb  = ws;                                      // M * HID
  bf16* Wqb = Xb  + M * HID_;                          // 2048*2048
  bf16* Wkb = Wqb + (size_t)NH_  * HD_ * HID_;         // 512*2048
  bf16* Wvb = Wkb + (size_t)NKV_ * HD_ * HID_;
  bf16* Wob = Wvb + (size_t)NKV_ * HD_ * HID_;         // 2048*2048
  bf16* Qb  = Wob + (size_t)HID_ * NH_ * HD_;          // M * 2048
  bf16* Kb  = Qb  + M * NH_  * HD_;                    // M * 512
  bf16* Vb  = Kb  + M * NKV_ * HD_;
  bf16* Vtb = Vb  + M * NKV_ * HD_;
  bf16* Ob  = Vtb + M * NKV_ * HD_;                    // M * 2048

  // fp32 -> bf16
  cvt_f32_bf16_kernel<<<2048, 256, 0, stream>>>(X,  Xb,  (long)M * HID_);
  cvt_f32_bf16_kernel<<<2048, 256, 0, stream>>>(Wq, Wqb, (long)NH_  * HD_ * HID_);
  cvt_f32_bf16_kernel<<<1024, 256, 0, stream>>>(Wk, Wkb, (long)NKV_ * HD_ * HID_);
  cvt_f32_bf16_kernel<<<1024, 256, 0, stream>>>(Wv, Wvb, (long)NKV_ * HD_ * HID_);
  cvt_f32_bf16_kernel<<<2048, 256, 0, stream>>>(Wo, Wob, (long)HID_ * NH_ * HD_);

  dim3 blk(256);
  // QKV projections (block tile 256x128)
  gemm_bf16_kernel<<<dim3((NH_  * HD_) / 128, M / 256), blk, 0, stream>>>(
      Xb, Wqb, nullptr, Qb, (int)M, NH_ * HD_, HID_);
  gemm_bf16_kernel<<<dim3((NKV_ * HD_) / 128, M / 256), blk, 0, stream>>>(
      Xb, Wkb, nullptr, Kb, (int)M, NKV_ * HD_, HID_);
  gemm_bf16_kernel<<<dim3((NKV_ * HD_) / 128, M / 256), blk, 0, stream>>>(
      Xb, Wvb, nullptr, Vb, (int)M, NKV_ * HD_, HID_);

  // RoPE (in place) + V transpose
  long ropeQ = (long)M * NH_  * 64;
  long ropeK = (long)M * NKV_ * 64;
  rope_kernel<<<(ropeQ + 255) / 256, 256, 0, stream>>>(Qb, sinT, cosT, NH_,  ropeQ);
  rope_kernel<<<(ropeK + 255) / 256, 256, 0, stream>>>(Kb, sinT, cosT, NKV_, ropeK);
  long vtn = (long)M * NKV_ * HD_;
  vtrans_kernel<<<(vtn + 255) / 256, 256, 0, stream>>>(Vb, Vtb);

  // flash attention: B*NH*(S/16) = 4096 wave-tiles, 8 waves per block
  flash_attn_kernel<<<(B_ * NH_ * (S_ / 16)) / 8, 256, 0, stream>>>(Qb, Kb, Vtb, Ob);

  // output projection -> fp32 d_out
  gemm_bf16_kernel<<<dim3(HID_ / 128, M / 256), blk, 0, stream>>>(
      Ob, Wob, (float*)d_out, nullptr, (int)M, HID_, HID_);
}
